// InstrumentedPSIModel_9062380995375
// MI455X (gfx1250) — compile-verified, matrix operand order checked by reading, not run on hardware
//
#include <hip/hip_runtime.h>
#include <hip/hip_bf16.h>

typedef __bf16 bf16;
typedef __attribute__((ext_vector_type(16))) __bf16 bf16x16;
typedef __attribute__((ext_vector_type(8)))  float  f32x8;
typedef int v4i __attribute__((vector_size(16)));   // 128-bit chunk type

#define BDIM   4
#define SEQ    2048
#define INDIM  64
#define DMODEL 1024
#define OUTDIM 256
#define NLAYER 4
#define NTOK   (BDIM * SEQ)        // 8192 tokens
#define NCHUNK 16
#define CHUNK  (SEQ / NCHUNK)      // 128

#define LDS_AS __attribute__((address_space(3)))
#define GLB_AS __attribute__((address_space(1)))

// ---------------------------------------------------------------------------
// Async global -> LDS 16-byte copy (CDNA5 GLOBAL_LOAD_ASYNC_TO_LDS_B128,
// tracked with ASYNCcnt). Builtin signature (probe-derived):
//   (v4i addrspace(1)* src, v4i addrspace(3)* dst, imm offset, imm cpol)
// ---------------------------------------------------------------------------
__device__ __forceinline__ void async_cp16(bf16* dst_lds, const bf16* src_glb) {
#if __has_builtin(__builtin_amdgcn_global_load_async_to_lds_b128)
  __builtin_amdgcn_global_load_async_to_lds_b128(
      (GLB_AS v4i*)src_glb, (LDS_AS v4i*)dst_lds, 0, 0);
#else
  asm volatile("global_load_async_to_lds_b128 %0, %1, off"
               :: "v"((LDS_AS v4i*)dst_lds), "v"((GLB_AS v4i*)src_glb)
               : "memory");
#endif
}

template <int N>
__device__ __forceinline__ void wait_async() {
#if __has_builtin(__builtin_amdgcn_s_wait_asynccnt)
  __builtin_amdgcn_s_wait_asynccnt(N);
#else
  asm volatile("s_wait_asynccnt %0" :: "i"(N) : "memory");
#endif
}

// ---------------------------------------------------------------------------
// bf16 WMMA GEMM: C[M,N] = A[M,K] * B[K,N], with B pre-transposed: BT[N][K].
// Tiles: BM=128, BN=128, BK=32; 8 waves, each owns a 32x64 patch (2x4 wmma
// tiles of v_wmma_f32_16x16x32_bf16). Double-buffered LDS fed by async
// global->LDS copies (4 async instrs / wave / tile, pipelined with
// s_wait_asynccnt 4).
// MODE 0: outF = acc + bias
// MODE 1: outH = gelu(acc + bias)
// MODE 2: outF = outF + addB + acc + bias   (residual: h += hn + mlp)
// Assumes M%128==0, N%128==0 or N==256, K%32==0 (true for all calls here).
// ---------------------------------------------------------------------------
#define BM  128
#define BN  128
#define BK  32
#define LDT 40   // padded LDS row stride (bf16); 80B keeps 16B alignment

__device__ __forceinline__ bf16x16 ld_frag(const bf16* base, int ld, int lane) {
  // 16-bit A/B fragment layout (ISA 7.12.2): lane = half*16 + r,
  // VGPR v holds k = (v<4?0:16) + half*8 + (v&3)*2 and k+1 (packed pair).
  const int r = lane & 15;
  const int half = lane >> 4;
  bf16x16 f;
#pragma unroll
  for (int v = 0; v < 8; ++v) {
    const int k = ((v & 4) ? 16 : 0) + half * 8 + (v & 3) * 2;
    f[2 * v]     = base[r * ld + k];
    f[2 * v + 1] = base[r * ld + k + 1];
  }
  return f;
}

template <int MODE>
__global__ __launch_bounds__(256) void gemm_bf16_wmma(
    const bf16* __restrict__ A, const bf16* __restrict__ BT,
    const float* __restrict__ bias,
    float* __restrict__ outF, bf16* __restrict__ outH,
    const float* __restrict__ addB,
    int M, int N, int K) {
  __shared__ bf16 As[2][BM * LDT];
  __shared__ bf16 Bs[2][BN * LDT];   // [n][k] (weights stored transposed)

  const int tid  = threadIdx.x;
  const int lane = tid & 31;
  const int wave = tid >> 5;
  const int wm = wave & 3;        // 0..3 : 32-row strip
  const int wn = wave >> 2;       // 0..1 : 64-col strip
  const int rowBase = blockIdx.y * BM;
  const int colBase = blockIdx.x * BN;

  f32x8 acc[2][4];
#pragma unroll
  for (int i = 0; i < 2; ++i)
#pragma unroll
    for (int j = 0; j < 4; ++j)
#pragma unroll
      for (int e = 0; e < 8; ++e) acc[i][j][e] = 0.0f;

  // Issue one tile's async copies: A and B tiles are both 128x32 bf16,
  // 512 chunks of 16B each -> 2 A-chunks + 2 B-chunks per thread.
  auto issue_tile = [&](int buf, int kt) {
#pragma unroll
    for (int c = 0; c < 2; ++c) {
      const int chunk = tid * 2 + c;        // 0..511
      const int r  = chunk >> 2;            // 0..127
      const int kc = (chunk & 3) * 8;       // 0,8,16,24
      async_cp16(&As[buf][r * LDT + kc], A  + (size_t)(rowBase + r) * K + kt + kc);
      async_cp16(&Bs[buf][r * LDT + kc], BT + (size_t)(colBase + r) * K + kt + kc);
    }
  };

  const int T = K / BK;
  issue_tile(0, 0);
  for (int t = 0; t < T; ++t) {
    const int cur = t & 1;
    if (t + 1 < T) {
      issue_tile(1 - cur, (t + 1) * BK);
      wait_async<4>();        // 4 newest (next tile) may remain in flight
    } else {
      wait_async<0>();
    }
    __syncthreads();

    bf16x16 af[2], bfr[4];
#pragma unroll
    for (int i = 0; i < 2; ++i)
      af[i] = ld_frag(&As[cur][(wm * 32 + i * 16) * LDT], LDT, lane);
#pragma unroll
    for (int j = 0; j < 4; ++j)
      bfr[j] = ld_frag(&Bs[cur][(wn * 64 + j * 16) * LDT], LDT, lane);
#pragma unroll
    for (int i = 0; i < 2; ++i)
#pragma unroll
      for (int j = 0; j < 4; ++j)
        acc[i][j] = __builtin_amdgcn_wmma_f32_16x16x32_bf16(
            false, af[i], false, bfr[j], (short)0, acc[i][j], false, false);
    __syncthreads();   // protect buffer reuse two iterations later
  }

  // ---- epilogue (C/D layout: VGPR r -> m = r + (lane>>4)*8, n = lane&15) ----
  const int ln = lane & 15;
  const int lh = lane >> 4;
#pragma unroll
  for (int i = 0; i < 2; ++i)
#pragma unroll
    for (int j = 0; j < 4; ++j)
#pragma unroll
      for (int r = 0; r < 8; ++r) {
        const int row = rowBase + wm * 32 + i * 16 + lh * 8 + r;
        const int col = colBase + wn * 64 + j * 16 + ln;
        const size_t idx = (size_t)row * N + col;
        const float v = acc[i][j][r] + bias[col];
        if (MODE == 0) {
          outF[idx] = v;
        } else if (MODE == 1) {
          outH[idx] = (bf16)(0.5f * v * (1.0f + erff(v * 0.70710678118f)));
        } else {
          outF[idx] = outF[idx] + addB[idx] + v;  // h += hn + mlp_out
        }
      }
}

// ---------------------------------------------------------------------------
// LayerNorm over F features, one block (256 thr) per token.
// ---------------------------------------------------------------------------
template <int F, bool WRITE_BF16, bool WRITE_F32, bool IN_BF16>
__global__ __launch_bounds__(256) void layernorm_k(
    const void* __restrict__ in, const float* __restrict__ w,
    const float* __restrict__ b, float* __restrict__ outF,
    bf16* __restrict__ outH) {
  const int tok = blockIdx.x;
  const int tid = threadIdx.x;
  constexpr int PER = F / 256;
  const float* inF = (const float*)in;
  const bf16*  inH = (const bf16*)in;

  float vals[PER];
  float s = 0.0f, sq = 0.0f;
#pragma unroll
  for (int i = 0; i < PER; ++i) {
    const int f = tid + i * 256;
    const float v = IN_BF16 ? (float)inH[(size_t)tok * F + f]
                            : inF[(size_t)tok * F + f];
    vals[i] = v;
    s += v;
    sq += v * v;
  }
#pragma unroll
  for (int off = 16; off > 0; off >>= 1) {
    s  += __shfl_down(s, off);
    sq += __shfl_down(sq, off);
  }
  __shared__ float red_s[8], red_q[8];
  const int lane = tid & 31, wv = tid >> 5;
  if (lane == 0) { red_s[wv] = s; red_q[wv] = sq; }
  __syncthreads();
  if (tid == 0) {
    float ts = 0.0f, tq = 0.0f;
#pragma unroll
    for (int i = 0; i < 8; ++i) { ts += red_s[i]; tq += red_q[i]; }
    red_s[0] = ts; red_q[0] = tq;
  }
  __syncthreads();
  const float mean = red_s[0] * (1.0f / F);
  const float var  = red_q[0] * (1.0f / F) - mean * mean;
  const float inv  = rsqrtf(var + 1e-5f);
#pragma unroll
  for (int i = 0; i < PER; ++i) {
    const int f = tid + i * 256;
    const float nv = (vals[i] - mean) * inv * w[f] + b[f];
    if (WRITE_F32)  outF[(size_t)tok * F + f] = nv;
    if (WRITE_BF16) outH[(size_t)tok * F + f] = (bf16)nv;
  }
}

// ---------------------------------------------------------------------------
// Embedding: h = x @ in_w + in_b + pos_encoding   (one thread per (tok,d))
// ---------------------------------------------------------------------------
__global__ __launch_bounds__(256) void embed_k(
    const float* __restrict__ x, const float* __restrict__ in_w,
    const float* __restrict__ in_b, float* __restrict__ h) {
  const int idx = blockIdx.x * 256 + threadIdx.x;  // NTOK*DMODEL
  const int d = idx % DMODEL;
  const int tok = idx / DMODEL;
  const int s = tok % SEQ;
  float acc = in_b[d];
  const float* xr = x + (size_t)tok * INDIM;
#pragma unroll 8
  for (int i = 0; i < INDIM; ++i) acc += xr[i] * in_w[i * DMODEL + d];
  const int d2 = d & ~1;
  const float div = expf((float)d2 * (-9.210340371976184f / (float)DMODEL));
  const float ang = (float)s * div;
  acc += (d & 1) ? cosf(ang) : sinf(ang);
  h[idx] = acc;
}

// Convert f32 [K][N] weight to bf16 transposed [N][K] (write-coalesced).
__global__ __launch_bounds__(256) void f32_to_bf16_tr_k(
    const float* __restrict__ in, bf16* __restrict__ out, int K, int N) {
  const int idx = blockIdx.x * 256 + threadIdx.x;   // over N*K (k fast)
  const int k = idx % K;
  const int n = idx / K;
  out[(size_t)n * K + k] = (bf16)in[(size_t)k * N + n];
}

// ---------------------------------------------------------------------------
// Chunked scan (16 chunks x 128 steps) over S per (b,d) channel.
// ---------------------------------------------------------------------------
__global__ __launch_bounds__(256) void scan_osum_k(
    const float* __restrict__ omega, const float* __restrict__ log_scale,
    float* __restrict__ osum) {
  const int idx = blockIdx.x * 256 + threadIdx.x;  // B*NCHUNK*DMODEL
  const int d = idx % DMODEL;
  const int c = (idx / DMODEL) % NCHUNK;
  const int b = idx / (DMODEL * NCHUNK);
  const float esc = expf(log_scale[d]);
  const int s0 = c * CHUNK;
  float s = 0.0f;
  for (int t = 0; t < CHUNK; ++t) {
    const int sg = s0 + t;
    s += omega[((size_t)b * SEQ + sg) * DMODEL + d] * esc *
         rsqrtf((float)(sg + 1));
  }
  osum[idx] = s;
}

__global__ __launch_bounds__(256) void chunk_prefix_k(
    const float* __restrict__ in1, float* __restrict__ out1,
    const float* __restrict__ in2, float* __restrict__ out2) {
  const int idx = blockIdx.x * 256 + threadIdx.x;  // B*DMODEL
  const int d = idx % DMODEL;
  const int b = idx / DMODEL;
  float r1 = 0.0f, r2 = 0.0f;
  for (int c = 0; c < NCHUNK; ++c) {
    const size_t o = ((size_t)b * NCHUNK + c) * DMODEL + d;
    out1[o] = r1; r1 += in1[o];
    if (in2 != nullptr) { out2[o] = r2; r2 += in2[o]; }
  }
}

__global__ __launch_bounds__(256) void scan_phase_k(
    const float* __restrict__ omega, const float* __restrict__ log_scale,
    const float* __restrict__ phioff, const float* __restrict__ hn,
    bf16* __restrict__ ctx, float* __restrict__ cosb, float* __restrict__ sinb,
    float* __restrict__ xcsum, float* __restrict__ xssum) {
  const int idx = blockIdx.x * 256 + threadIdx.x;
  const int d = idx % DMODEL;
  const int c = (idx / DMODEL) % NCHUNK;
  const int b = idx / (DMODEL * NCHUNK);
  const float esc = expf(log_scale[d]);
  float phi = phioff[idx];
  float sc = 0.0f, ss = 0.0f;
  const int s0 = c * CHUNK;
  for (int t = 0; t < CHUNK; ++t) {
    const int sg = s0 + t;
    const size_t tok = (size_t)b * SEQ + sg;
    phi += omega[tok * DMODEL + d] * esc * rsqrtf((float)(sg + 1));
    const float cp = cosf(phi), sp = sinf(phi);
    const float x = hn[tok * DMODEL + d];
    const float xc = x * cp, xs = x * sp;
    ctx[tok * 4 * DMODEL + d]          = (bf16)xc;
    ctx[tok * 4 * DMODEL + DMODEL + d] = (bf16)xs;
    cosb[tok * DMODEL + d] = cp;
    sinb[tok * DMODEL + d] = sp;
    sc += xc; ss += xs;
  }
  xcsum[idx] = sc;
  xssum[idx] = ss;
}

__global__ __launch_bounds__(256) void scan_ret_k(
    const float* __restrict__ hn, const float* __restrict__ cosb,
    const float* __restrict__ sinb, const float* __restrict__ xcoff,
    const float* __restrict__ xsoff, bf16* __restrict__ ctx) {
  const int idx = blockIdx.x * 256 + threadIdx.x;
  const int d = idx % DMODEL;
  const int c = (idx / DMODEL) % NCHUNK;
  const int b = idx / (DMODEL * NCHUNK);
  float rc = xcoff[idx], rs = xsoff[idx];
  const int s0 = c * CHUNK;
  for (int t = 0; t < CHUNK; ++t) {
    const int sg = s0 + t;
    const size_t tok = (size_t)b * SEQ + sg;
    const float cp = cosb[tok * DMODEL + d];
    const float sp = sinb[tok * DMODEL + d];
    const float x = hn[tok * DMODEL + d];
    rc += x * cp; rs += x * sp;
    const float inv = 1.0f / (float)(sg + 1);
    const float mr = rc * inv, mi = rs * inv;
    ctx[tok * 4 * DMODEL + 2 * DMODEL + d] = (bf16)(mr * cp + mi * sp);
    ctx[tok * 4 * DMODEL + 3 * DMODEL + d] = (bf16)(mi * cp - mr * sp);
  }
}

// ---------------------------------------------------------------------------
extern "C" void kernel_launch(void* const* d_in, const int* in_sizes, int n_in,
                              void* d_out, int out_size, void* d_ws,
                              size_t ws_size, hipStream_t stream) {
  const float* x        = (const float*)d_in[0];
  const float* in_w     = (const float*)d_in[1];
  const float* in_b     = (const float*)d_in[2];
  const float* norm_w   = (const float*)d_in[3];
  const float* norm_b   = (const float*)d_in[4];
  const float* omega_w  = (const float*)d_in[5];
  const float* omega_b  = (const float*)d_in[6];
  const float* log_sc   = (const float*)d_in[7];
  const float* ln4_w    = (const float*)d_in[8];
  const float* ln4_b    = (const float*)d_in[9];
  const float* w1       = (const float*)d_in[10];
  const float* b1       = (const float*)d_in[11];
  const float* w2       = (const float*)d_in[12];
  const float* b2       = (const float*)d_in[13];
  const float* fnorm_w  = (const float*)d_in[14];
  const float* fnorm_b  = (const float*)d_in[15];
  const float* out_w    = (const float*)d_in[16];
  const float* out_b    = (const float*)d_in[17];

  char* p = (char*)d_ws;
  auto take = [&](size_t bytes) -> void* {
    void* q = (void*)p;
    p += (bytes + 255) & ~(size_t)255;
    return q;
  };
  float* hbuf   = (float*)take((size_t)NTOK * DMODEL * 4);
  float* hnf    = (float*)take((size_t)NTOK * DMODEL * 4);
  bf16*  hnh    = (bf16*) take((size_t)NTOK * DMODEL * 2);
  float* omega  = (float*)take((size_t)NTOK * DMODEL * 4);
  float* cosb   = (float*)take((size_t)NTOK * DMODEL * 4);
  float* sinb   = (float*)take((size_t)NTOK * DMODEL * 4);
  bf16*  ctx    = (bf16*) take((size_t)NTOK * 4 * DMODEL * 2);
  bf16*  h1     = (bf16*) take((size_t)NTOK * 2 * DMODEL * 2);
  float* osum   = (float*)take((size_t)BDIM * NCHUNK * DMODEL * 4);
  float* phioff = (float*)take((size_t)BDIM * NCHUNK * DMODEL * 4);
  float* xcsum  = (float*)take((size_t)BDIM * NCHUNK * DMODEL * 4);
  float* xssum  = (float*)take((size_t)BDIM * NCHUNK * DMODEL * 4);
  float* xcoff  = (float*)take((size_t)BDIM * NCHUNK * DMODEL * 4);
  float* xsoff  = (float*)take((size_t)BDIM * NCHUNK * DMODEL * 4);
  bf16*  wObT   = (bf16*) take((size_t)NLAYER * DMODEL * DMODEL * 2);
  bf16*  w1bT   = (bf16*) take((size_t)NLAYER * 4 * DMODEL * 2 * DMODEL * 2);
  bf16*  w2bT   = (bf16*) take((size_t)NLAYER * 2 * DMODEL * DMODEL * 2);
  bf16*  owbT   = (bf16*) take((size_t)DMODEL * OUTDIM * 2);

  // --- convert weights to bf16 AND transpose to [N][K] (one pass per call;
  //     88MB of bf16 weights then sit in the 192MB L2 for all GEMMs) ---
  for (int l = 0; l < NLAYER; ++l) {
    int n;
    n = DMODEL * DMODEL;
    f32_to_bf16_tr_k<<<n / 256, 256, 0, stream>>>(
        omega_w + (size_t)l * n, wObT + (size_t)l * n, DMODEL, DMODEL);
    n = 4 * DMODEL * 2 * DMODEL;
    f32_to_bf16_tr_k<<<n / 256, 256, 0, stream>>>(
        w1 + (size_t)l * n, w1bT + (size_t)l * n, 4 * DMODEL, 2 * DMODEL);
    n = 2 * DMODEL * DMODEL;
    f32_to_bf16_tr_k<<<n / 256, 256, 0, stream>>>(
        w2 + (size_t)l * n, w2bT + (size_t)l * n, 2 * DMODEL, DMODEL);
  }
  f32_to_bf16_tr_k<<<(DMODEL * OUTDIM) / 256, 256, 0, stream>>>(
      out_w, owbT, DMODEL, OUTDIM);

  embed_k<<<(NTOK * DMODEL) / 256, 256, 0, stream>>>(x, in_w, in_b, hbuf);

  const int nsc = BDIM * NCHUNK * DMODEL;   // 65536 scan threads
  for (int l = 0; l < NLAYER; ++l) {
    layernorm_k<DMODEL, true, true, false><<<NTOK, 256, 0, stream>>>(
        hbuf, norm_w + l * DMODEL, norm_b + l * DMODEL, hnf, hnh);

    dim3 g0(DMODEL / BN, NTOK / BM);
    gemm_bf16_wmma<0><<<g0, 256, 0, stream>>>(
        hnh, wObT + (size_t)l * DMODEL * DMODEL, omega_b + l * DMODEL,
        omega, nullptr, nullptr, NTOK, DMODEL, DMODEL);

    scan_osum_k<<<nsc / 256, 256, 0, stream>>>(omega, log_sc + l * DMODEL, osum);
    chunk_prefix_k<<<(BDIM * DMODEL) / 256, 256, 0, stream>>>(
        osum, phioff, nullptr, nullptr);
    scan_phase_k<<<nsc / 256, 256, 0, stream>>>(
        omega, log_sc + l * DMODEL, phioff, hnf, ctx, cosb, sinb, xcsum, xssum);
    chunk_prefix_k<<<(BDIM * DMODEL) / 256, 256, 0, stream>>>(
        xcsum, xcoff, xssum, xsoff);
    scan_ret_k<<<nsc / 256, 256, 0, stream>>>(hnf, cosb, sinb, xcoff, xsoff, ctx);

    layernorm_k<4 * DMODEL, true, false, true><<<NTOK, 256, 0, stream>>>(
        ctx, ln4_w + l * 4 * DMODEL, ln4_b + l * 4 * DMODEL, nullptr, ctx);

    dim3 g1((2 * DMODEL) / BN, NTOK / BM);
    gemm_bf16_wmma<1><<<g1, 256, 0, stream>>>(
        ctx, w1bT + (size_t)l * 4 * DMODEL * 2 * DMODEL, b1 + l * 2 * DMODEL,
        nullptr, h1, nullptr, NTOK, 2 * DMODEL, 4 * DMODEL);

    dim3 g2(DMODEL / BN, NTOK / BM);
    gemm_bf16_wmma<2><<<g2, 256, 0, stream>>>(
        h1, w2bT + (size_t)l * 2 * DMODEL * DMODEL, b2 + l * DMODEL,
        hbuf, nullptr, hnf, NTOK, DMODEL, 2 * DMODEL);
  }

  layernorm_k<DMODEL, true, false, false><<<NTOK, 256, 0, stream>>>(
      hbuf, fnorm_w, fnorm_b, nullptr, hnh);

  dim3 gf(OUTDIM / BN, NTOK / BM);
  gemm_bf16_wmma<0><<<gf, 256, 0, stream>>>(
      hnh, owbT, out_b, (float*)d_out, nullptr, nullptr, NTOK, OUTDIM, DMODEL);
}
